// SPR_RGCN_LastToken_88648124989976
// MI455X (gfx1250) — matrix operand order repaired and checked by reading, not compiled, not run on hardware
//
#include <hip/hip_runtime.h>
#include <hip/hip_bf16.h>

typedef __attribute__((ext_vector_type(16))) _Float16 v16h;
typedef __attribute__((ext_vector_type(8)))  _Float16 v8h;
typedef __attribute__((ext_vector_type(8)))  float    v8f;

// ---------------------------------------------------------------------------
// Degree count per (relation, dst-node), graph-constant.
__global__ __launch_bounds__(256) void k_deg(const int* __restrict__ et,
                                             const int* __restrict__ dst,
                                             float* __restrict__ deg, int E, int N) {
    size_t tid = (size_t)blockIdx.x * 256 + threadIdx.x;
    if (tid >= (size_t)E) return;
    atomicAdd(&deg[(size_t)et[tid] * N + dst[tid]], 1.0f);
}

__global__ __launch_bounds__(256) void k_recip(float* __restrict__ deg, int n) {
    size_t tid = (size_t)blockIdx.x * 256 + threadIdx.x;
    if (tid >= (size_t)n) return;
    deg[tid] = 1.0f / fmaxf(deg[tid], 1.0f);
}

// ---------------------------------------------------------------------------
// Pack [W_root; W_r0; W_r1; W_r2] (rows = 4*Fin, cols = Fout) to f16.
__global__ __launch_bounds__(256) void k_pack_w(const float* __restrict__ root,
                                                const float* __restrict__ rel,
                                                _Float16* __restrict__ Wc,
                                                int Fin, int Fout) {
    int tot = 4 * Fin * Fout;
    int tid = blockIdx.x * 256 + threadIdx.x;
    if (tid >= tot) return;
    int kk = tid / Fout;
    int j  = tid - kk * Fout;
    float v;
    if (kk < Fin) {
        v = root[kk * Fout + j];
    } else {
        int t2 = kk - Fin;
        int r  = t2 / Fin;
        int k2 = t2 - r * Fin;
        v = rel[((size_t)r * Fin + k2) * Fout + j];
    }
    Wc[tid] = (_Float16)v;
}

// ---------------------------------------------------------------------------
// Node embedding + pre-linear (16->32) + ReLU, writes h block of X1 (f16).
__global__ __launch_bounds__(256) void k_embed(const int* __restrict__ x,
                                               const float* __restrict__ se,
                                               const float* __restrict__ ce,
                                               const float* __restrict__ pw,
                                               const float* __restrict__ pb,
                                               _Float16* __restrict__ X1, int N) {
    __shared__ float spw[16 * 32];
    __shared__ float semb[128];
    __shared__ float cemb[128];
    __shared__ float spb[32];
    int t = threadIdx.x;
    for (int i = t; i < 512; i += 256) spw[i] = pw[i];
    if (t < 128) { semb[t] = se[t]; cemb[t] = ce[t]; }
    if (t < 32) spb[t] = pb[t];
    __syncthreads();
    int n = blockIdx.x * 256 + t;
    if (n >= N) return;
    int si = x[2 * n], ci = x[2 * n + 1];
    float f[16];
#pragma unroll
    for (int i = 0; i < 8; i++) { f[i] = semb[si * 8 + i]; f[8 + i] = cemb[ci * 8 + i]; }
    _Float16* outp = X1 + (size_t)n * 128;
#pragma unroll
    for (int j = 0; j < 32; j++) {
        float acc = spb[j];
#pragma unroll
        for (int i = 0; i < 16; i++) acc += f[i] * spw[i * 32 + j];
        outp[j] = (_Float16)fmaxf(acc, 0.0f);
    }
}

// ---------------------------------------------------------------------------
// Scatter h[src] into per-relation accumulator (fp32 atomics), one relation r.
__global__ __launch_bounds__(256) void k_scatter(const int* __restrict__ src,
                                                 const int* __restrict__ dst,
                                                 const int* __restrict__ et,
                                                 const _Float16* __restrict__ Xh, int ldx,
                                                 float* __restrict__ acc,
                                                 int r, int logF, int E) {
    size_t tid = (size_t)blockIdx.x * 256 + threadIdx.x;
    size_t e = tid >> logF;
    int f = (int)(tid & ((1u << logF) - 1u));
    if (e >= (size_t)E) return;
    if (et[e] != r) return;
    float v = (float)Xh[(size_t)src[e] * ldx + f];
    atomicAdd(&acc[((size_t)dst[e] << logF) + f], v);
}

// Normalize accumulator by 1/max(deg,1) and write f16 into X agg block.
__global__ __launch_bounds__(256) void k_normalize(const float* __restrict__ acc,
                                                   const float* __restrict__ inv,
                                                   _Float16* __restrict__ X, int ldx,
                                                   int off, int logF, int N) {
    size_t tid = (size_t)blockIdx.x * 256 + threadIdx.x;
    size_t n = tid >> logF;
    int f = (int)(tid & ((1u << logF) - 1u));
    if (n >= (size_t)N) return;
    X[n * ldx + off + f] = (_Float16)(acc[tid] * inv[n]);
}

// ---------------------------------------------------------------------------
// Row-parallel GEMM: Out[N,64] = relu(X[N,K] @ W[K,64] + bias), K in {128,256}.
// 8 waves/block, each wave: 16 rows x 64 cols via 4x v_wmma_f32_16x16x32_f16.
__global__ __launch_bounds__(256) void k_gemm(const _Float16* __restrict__ X,
                                              const _Float16* __restrict__ W,
                                              const float* __restrict__ bias,
                                              _Float16* __restrict__ Out,
                                              int ldo, int N, int K) {
    __shared__ _Float16 sW[256 * 64];  // full Wcat, up to 32 KB
    const int tid = threadIdx.x;
    {   // cooperative stage of W into LDS (dword granularity)
        const uint32_t* wsrc = (const uint32_t*)W;
        uint32_t* wdst = (uint32_t*)sW;
        const int nW = K * 32;  // K*64 halves
        for (int i = tid; i < nW; i += 256) wdst[i] = wsrc[i];
    }
    __syncthreads();

    const int wave = tid >> 5;
    const int lane = tid & 31;
    const int r0 = blockIdx.x * 128 + wave * 16;
    if (r0 >= N) return;  // whole wave drops together; EXEC all-ones for WMMA

    // A fragment addressing: lanes 0-15 row M=lane, K = {0..7,16..23};
    // lanes 16-31 same rows, K = {8..15,24..31}.
    const int arow = r0 + (lane & 15);
    const int koff = (lane >> 4) * 8;
    const _Float16* rowp = X + (size_t)arow * K;

    v8f c[4];
#pragma unroll
    for (int i = 0; i < 4; i++) c[i] = (v8f){};

    for (int k0 = 0; k0 < K; k0 += 32) {
        v8h alo = *(const v8h*)(rowp + k0 + koff);
        v8h ahi = *(const v8h*)(rowp + k0 + 16 + koff);
        v16h a = __builtin_shufflevector(alo, ahi, 0, 1, 2, 3, 4, 5, 6, 7,
                                         8, 9, 10, 11, 12, 13, 14, 15);
        // B fragment: lane l holds column N=l%16, K in same order as A
        const _Float16* wr = sW + (size_t)(k0 + koff) * 64 + (lane & 15);
#pragma unroll
        for (int nt = 0; nt < 4; nt++) {
            v16h b;
#pragma unroll
            for (int s = 0; s < 16; s++)
                b[s] = wr[(size_t)(s < 8 ? s : s + 8) * 64 + nt * 16];
            c[nt] = __builtin_amdgcn_wmma_f32_16x16x32_f16(
                false, a, false, b, (short)0, c[nt], false, false);
        }
    }

    // C layout: VGPR rr, lanes 0-15 => M=rr, N=lane; lanes 16-31 => M=rr+8.
    const int mhi = (lane >> 4) * 8;
    const int ncol = lane & 15;
#pragma unroll
    for (int nt = 0; nt < 4; nt++) {
        const int n = nt * 16 + ncol;
        const float bv = bias[n];
#pragma unroll
        for (int rr = 0; rr < 8; rr++) {
            float v = c[nt][rr] + bv;
            v = v > 0.0f ? v : 0.0f;
            Out[(size_t)(r0 + mhi + rr) * ldo + n] = (_Float16)v;
        }
    }
}

// ---------------------------------------------------------------------------
// Classifier: out[g, c] = h2[ptr[g+1]-1] @ cls_w + cls_b
__global__ __launch_bounds__(256) void k_cls(const _Float16* __restrict__ h2,
                                             const int* __restrict__ ptr,
                                             const float* __restrict__ cw,
                                             const float* __restrict__ cb,
                                             float* __restrict__ out, int G) {
    int tid = blockIdx.x * 256 + threadIdx.x;
    int g = tid >> 4, c = tid & 15;
    if (g >= G || c >= 10) return;
    int idx = ptr[g + 1] - 1;
    const _Float16* hp = h2 + (size_t)idx * 64;
    float acc = cb[c];
#pragma unroll
    for (int k = 0; k < 64; k++) acc += (float)hp[k] * cw[k * 10 + c];
    out[g * 10 + c] = acc;
}

// ---------------------------------------------------------------------------
extern "C" void kernel_launch(void* const* d_in, const int* in_sizes, int n_in,
                              void* d_out, int out_size, void* d_ws, size_t ws_size,
                              hipStream_t stream) {
    const int* x    = (const int*)d_in[0];
    const int* ei   = (const int*)d_in[1];
    const int* et   = (const int*)d_in[2];
    const int* ptr  = (const int*)d_in[3];
    const float* se = (const float*)d_in[4];
    const float* ce = (const float*)d_in[5];
    const float* pw = (const float*)d_in[6];
    const float* pb = (const float*)d_in[7];
    const float* w1rel  = (const float*)d_in[8];
    const float* w1root = (const float*)d_in[9];
    const float* b1     = (const float*)d_in[10];
    const float* w2rel  = (const float*)d_in[11];
    const float* w2root = (const float*)d_in[12];
    const float* b2     = (const float*)d_in[13];
    const float* cw     = (const float*)d_in[14];
    const float* cb     = (const float*)d_in[15];
    float* out = (float*)d_out;

    const int N = in_sizes[0] / 2;
    const int E = in_sizes[2];
    const int G = in_sizes[3] - 1;
    const int* src = ei;
    const int* dst = ei + E;

    // Workspace layout (~582 MB)
    char* p = (char*)d_ws;
    auto take = [&](size_t bytes) -> char* {
        char* q = p;
        p += (bytes + 255) & ~(size_t)255;
        return q;
    };
    _Float16* X1  = (_Float16*)take((size_t)N * 128 * 2);  // [h32|agg0|agg1|agg2]
    _Float16* X2  = (_Float16*)take((size_t)N * 256 * 2);  // [h64|agg0|agg1|agg2]
    _Float16* h2  = (_Float16*)take((size_t)N * 64 * 2);
    float*    acc = (float*)take((size_t)N * 64 * 4);      // per-relation accumulator
    float*    deg = (float*)take((size_t)3 * N * 4);
    _Float16* Wc1 = (_Float16*)take(128 * 64 * 2);
    _Float16* Wc2 = (_Float16*)take(256 * 64 * 2);
    (void)ws_size; (void)n_in; (void)out_size;

    // Graph-constant reciprocal degrees
    hipMemsetAsync(deg, 0, (size_t)3 * N * 4, stream);
    k_deg<<<(E + 255) / 256, 256, 0, stream>>>(et, dst, deg, E, N);
    k_recip<<<(3 * N + 255) / 256, 256, 0, stream>>>(deg, 3 * N);

    // Weight packing to f16 Wcat layouts
    k_pack_w<<<(128 * 64 + 255) / 256, 256, 0, stream>>>(w1root, w1rel, Wc1, 32, 64);
    k_pack_w<<<(256 * 64 + 255) / 256, 256, 0, stream>>>(w2root, w2rel, Wc2, 64, 64);

    // Embedding + pre-linear -> X1 h block
    k_embed<<<(N + 255) / 256, 256, 0, stream>>>(x, se, ce, pw, pb, X1, N);

    // Layer 1 aggregation (F=32), one relation at a time to keep acc small
    for (int r = 0; r < 3; r++) {
        hipMemsetAsync(acc, 0, (size_t)N * 32 * 4, stream);
        size_t tot = (size_t)E * 32;
        k_scatter<<<(unsigned)((tot + 255) / 256), 256, 0, stream>>>(
            src, dst, et, X1, 128, acc, r, 5, E);
        size_t tn = (size_t)N * 32;
        k_normalize<<<(unsigned)((tn + 255) / 256), 256, 0, stream>>>(
            acc, deg + (size_t)r * N, X1, 128, (1 + r) * 32, 5, N);
    }
    // Layer 1 fused GEMM: X1[N,128] @ Wc1[128,64] -> relu -> X2 h block
    k_gemm<<<(N + 127) / 128, 256, 0, stream>>>(X1, Wc1, b1, X2, 256, N, 128);

    // Layer 2 aggregation (F=64)
    for (int r = 0; r < 3; r++) {
        hipMemsetAsync(acc, 0, (size_t)N * 64 * 4, stream);
        size_t tot = (size_t)E * 64;
        k_scatter<<<(unsigned)((tot + 255) / 256), 256, 0, stream>>>(
            src, dst, et, X2, 256, acc, r, 6, E);
        size_t tn = (size_t)N * 64;
        k_normalize<<<(unsigned)((tn + 255) / 256), 256, 0, stream>>>(
            acc, deg + (size_t)r * N, X2, 256, (1 + r) * 64, 6, N);
    }
    // Layer 2 fused GEMM: X2[N,256] @ Wc2[256,64] -> relu -> h2
    k_gemm<<<(N + 127) / 128, 256, 0, stream>>>(X2, Wc2, b2, h2, 64, N, 256);

    // Readout
    k_cls<<<(G * 16 + 255) / 256, 256, 0, stream>>>(h2, ptr, cw, cb, out, G);
}
